// CausalSelfAttention_55044300865695
// MI455X (gfx1250) — matrix operand order repaired and checked by reading, not compiled, not run on hardware
//
#include <hip/hip_runtime.h>
#include <hip/hip_bf16.h>

// Problem constants (B=2, T=2048, C=1024, H=16, hd=64)
#define TVAL 2048
#define CVAL 1024
#define HVAL 16
#define HD   64
#define BTOT 4096   // B*T

typedef unsigned short u16;
typedef __attribute__((ext_vector_type(16))) __bf16 v16bf;
typedef __attribute__((ext_vector_type(8)))  float  v8f;
typedef __attribute__((ext_vector_type(8)))  u16    ushort8;
typedef __attribute__((ext_vector_type(4)))  unsigned uint4v;
typedef __attribute__((ext_vector_type(8)))  int      int8v;
typedef __attribute__((ext_vector_type(4)))  int      int4v;

union FragA { v16bf bf; ushort8 u[2]; };

__device__ __forceinline__ u16 f2bf(float f) {
  unsigned u = __float_as_uint(f);
  u += 0x7FFFu + ((u >> 16) & 1u);   // round-to-nearest-even
  return (u16)(u >> 16);
}

// LDS aperture maps flat addr[31:0] directly to the LDS byte offset
// (CDNA5 ISA 10.2).
__device__ __forceinline__ unsigned ldsoff(const void* p) {
  return (unsigned)(size_t)p;
}

// CDNA5 async copy: global -> LDS, 16 bytes per lane, tracked by ASYNCcnt.
__device__ __forceinline__ void async_ld16(unsigned lds, const u16* g) {
  asm volatile("global_load_async_to_lds_b128 %0, %1, off"
               :: "v"(lds), "v"(g) : "memory");
}
__device__ __forceinline__ void wait_async0() {
  asm volatile("s_wait_asynccnt 0" ::: "memory");
}

// CDNA5 Tensor Data Mover: DMA a 2D tile (bf16 elements) global -> LDS.
// Descriptor per ISA ch.8: group0 {count=1, lds_addr, global_addr, type=2},
// group1 {data_size=1(2B), tensor_dim0/1, tile_dim0/1, dim0_stride}.
__device__ __forceinline__ void tdm_load_2d(unsigned lds_addr, const void* gp,
                                            unsigned dim0, unsigned dim1,
                                            unsigned tile0, unsigned tile1,
                                            unsigned stride0) {
  const unsigned long long ga = (unsigned long long)gp;
  uint4v g0;
  g0[0] = 1u;                                   // count=1, user D#
  g0[1] = lds_addr;                             // LDS byte address
  g0[2] = (unsigned)ga;                         // global_addr[31:0]
  g0[3] = (unsigned)(ga >> 32) | (2u << 30);    // addr[56:32] | type=2
  int8v g1;
  g1[0] = (int)(1u << 16);                      // data_size=1 (2 bytes)
  g1[1] = (int)(dim0 << 16);                    // tensor_dim0[15:0]
  g1[2] = (int)((dim0 >> 16) | (dim1 << 16));   // dim0[31:16] | dim1[15:0]
  g1[3] = (int)((dim1 >> 16) | (tile0 << 16));  // dim1[31:16] | tile_dim0
  g1[4] = (int)tile1;                           // tile_dim1; tile_dim2=0
  g1[5] = (int)stride0;                         // tensor_dim0_stride[31:0]
  g1[6] = 0;
  g1[7] = 0;
  int4v z4 = {0, 0, 0, 0};
#if defined(__clang_major__) && (__clang_major__ >= 23)
  int8v z8 = {0, 0, 0, 0, 0, 0, 0, 0};
  __builtin_amdgcn_tensor_load_to_lds(g0, g1, z4, z4, z8, 0);
#else
  __builtin_amdgcn_tensor_load_to_lds(g0, g1, z4, z4, 0);
#endif
}

// ---------------------------------------------------------------- convert
__global__ void f32_to_bf16_kernel(const float* __restrict__ in,
                                   u16* __restrict__ out, int n) {
  int i = blockIdx.x * blockDim.x + threadIdx.x;
  int stride = gridDim.x * blockDim.x;
  for (; i < n; i += stride) out[i] = f2bf(in[i]);
}

// ---------------------------------------------------------------- GEMM
// out[m,n] = sum_k A[m,k] * W[n,k] + bias[n]
// MODE 0: store fp32 to outF[m*N+n]   (output projection)
// MODE 1: scatter bf16 into q/k/v head layout [B,H,T,hd]   (QKV projection)
template <int MODE>
__global__ __launch_bounds__(128)
void gemm_bf16_wmma(const u16* __restrict__ A, const u16* __restrict__ W,
                    const float* __restrict__ bias, float* __restrict__ outF,
                    u16* __restrict__ Qo, u16* __restrict__ Ko,
                    u16* __restrict__ Vo, int M, int N, int K) {
  __shared__ u16 As[64 * 40];   // 64 rows x 32 K (+8 pad)
  __shared__ u16 Bs[64 * 40];

  const int tid   = threadIdx.x;
  const int lane  = tid & 31;
  const int w     = tid >> 5;
  const int lhalf = lane & 15;
  const int hi    = lane >> 4;
  const int m0 = blockIdx.y * 64;
  const int n0 = blockIdx.x * 64;
  const int wm = (w >> 1) * 32;   // wave 32x32 tile origin
  const int wn = (w & 1) * 32;

  v8f acc[2][2];
#pragma unroll
  for (int i = 0; i < 2; ++i)
#pragma unroll
    for (int j = 0; j < 2; ++j) acc[i][j] = 0.0f;

  const int lrow = tid >> 1;          // 0..63
  const int loff = (tid & 1) * 16;    // 0 or 16
  const unsigned asO = ldsoff(&As[lrow * 40 + loff]);
  const unsigned bsO = ldsoff(&Bs[lrow * 40 + loff]);

  for (int k0 = 0; k0 < K; k0 += 32) {
    // cooperative async stage of A (64x32) and B (64x32) tiles into LDS
    const u16* ga = A + (size_t)(m0 + lrow) * K + k0 + loff;
    const u16* gb = W + (size_t)(n0 + lrow) * K + k0 + loff;
    async_ld16(asO,      ga);
    async_ld16(asO + 16, ga + 8);
    async_ld16(bsO,      gb);
    async_ld16(bsO + 16, gb + 8);
    if (k0 + 32 < K) {               // hint next tile toward the caches
      __builtin_prefetch(ga + 32, 0, 0);
      __builtin_prefetch(gb + 32, 0, 0);
    }
    wait_async0();
    __syncthreads();

    // A-fragment (16x32 bf16): lane<16 -> M=lane, K 0..7 & 16..23
    //                          lane>=16 -> M=lane-16, K 8..15 & 24..31
    FragA a[2], b[2];
#pragma unroll
    for (int i = 0; i < 2; ++i) {
      const int row = wm + i * 16 + lhalf;
      const int ko  = hi * 8;
      a[i].u[0] = *(const ushort8*)&As[row * 40 + ko];
      a[i].u[1] = *(const ushort8*)&As[row * 40 + ko + 16];
    }
    // B-fragment (32x16 bf16): lane<16 -> N=lane, K 0..7 & 16..23 ; lane>=16 -> K 8..15 & 24..31
#pragma unroll
    for (int j = 0; j < 2; ++j) {
      const int row = wn + j * 16 + lhalf;
      const int ko  = hi * 8;
      b[j].u[0] = *(const ushort8*)&Bs[row * 40 + ko];
      b[j].u[1] = *(const ushort8*)&Bs[row * 40 + ko + 16];
    }
#pragma unroll
    for (int i = 0; i < 2; ++i)
#pragma unroll
      for (int j = 0; j < 2; ++j)
        acc[i][j] = __builtin_amdgcn_wmma_f32_16x16x32_bf16(
            false, a[i].bf, false, b[j].bf, (short)0, acc[i][j], false, false);
    __syncthreads();
  }

  // epilogue: C/D layout -> lane<16 rows r, lane>=16 rows r+8, col = lane&15
#pragma unroll
  for (int i = 0; i < 2; ++i) {
#pragma unroll
    for (int j = 0; j < 2; ++j) {
#pragma unroll
      for (int r = 0; r < 8; ++r) {
        const int m = m0 + wm + i * 16 + r + hi * 8;
        const int n = n0 + wn + j * 16 + lhalf;
        const float v = acc[i][j][r] + bias[n];
        if (MODE == 0) {
          outF[(size_t)m * N + n] = v;
        } else {
          const int part = n >> 10;      // 0=q 1=k 2=v
          const int oc   = n & 1023;
          const int hh   = oc >> 6;
          const int d    = oc & 63;
          const int bb   = m >> 11;      // m = b*T + t
          const int t    = m & 2047;
          const size_t idx =
              ((size_t)((bb * HVAL + hh) * TVAL + t)) * HD + d;
          const u16 hv = f2bf(v);
          if (part == 0)      Qo[idx] = hv;
          else if (part == 1) Ko[idx] = hv;
          else                Vo[idx] = hv;
        }
      }
    }
  }
}

// ---------------------------------------------------------------- attention
// Flash-style causal attention per (b,h). Block = 64 query rows, 4 waves,
// each wave owns 16 query rows. Key blocks of 32; K tile DMA'd by the TDM,
// V tile transposed into LDS by VALU lanes.
__global__ __launch_bounds__(128)
void attn_wmma(const u16* __restrict__ Qg, const u16* __restrict__ Kg,
               const u16* __restrict__ Vg, u16* __restrict__ Yg) {
  __shared__ u16 Kt[32 * 64];        // K tile row-major [32 keys][64 d] (TDM)
  __shared__ u16 Vt[64 * 40];        // V tile transposed [64 d][32 keys]+pad
  __shared__ u16 Pb[4 * 16 * 32];    // per-wave P bounce buffer

  const int tid   = threadIdx.x;
  const int lane  = tid & 31;
  const int w     = tid >> 5;
  const int lhalf = lane & 15;
  const int hi    = lane >> 4;
  const int bh = blockIdx.y;              // b*H + h
  const int qb = blockIdx.x;              // 64-row query block
  const size_t base = (size_t)bh * TVAL * HD;
  const int q0 = qb * 64 + w * 16;        // this wave's first query row

  // Q tile 16x64 in registers as two 16x32 A-fragments
  FragA qf[2];
#pragma unroll
  for (int ks = 0; ks < 2; ++ks) {
    const u16* gq = Qg + base + (size_t)(q0 + lhalf) * HD + ks * 32 + hi * 8;
    qf[ks].u[0] = *(const ushort8*)(gq);
    qf[ks].u[1] = *(const ushort8*)(gq + 16);
  }

  v8f o[4];
#pragma unroll
  for (int dt = 0; dt < 4; ++dt) o[dt] = 0.0f;
  float mr[8], sr[8];
#pragma unroll
  for (int r = 0; r < 8; ++r) { mr[r] = -1e30f; sr[r] = 0.0f; }

  const int nKB  = 2 * qb + 2;          // causal: keys up to end of block
  const int lrow = tid >> 2;            // 0..31 (key row within tile)
  const int loff = (tid & 3) * 16;      // 0,16,32,48 (d offset)
  u16* pw = &Pb[w * 512];
  const unsigned ktO = ldsoff(&Kt[0]);

  for (int jb = 0; jb < nKB; ++jb) {
    const int kb = jb * 32;
    // --- stage K tile via Tensor Data Mover (one DMA for 32x64 bf16) ---
    if (w == 0) {
      tdm_load_2d(ktO, Kg + base + (size_t)kb * HD,
                  /*dim0=*/HD, /*dim1=*/TVAL,
                  /*tile0=*/HD, /*tile1=*/32, /*stride0=*/HD);
    }
    // --- stage V tile transposed (lanes scatter halfwords) ---
    const u16* gv = Vg + base + (size_t)(kb + lrow) * HD + loff;
    const ushort8 v0v = *(const ushort8*)(gv);
    const ushort8 v1v = *(const ushort8*)(gv + 8);
#pragma unroll
    for (int j = 0; j < 8; ++j) {
      Vt[(loff + j) * 40 + lrow]     = v0v[j];
      Vt[(loff + 8 + j) * 40 + lrow] = v1v[j];
    }
    if (w == 0) __builtin_amdgcn_s_wait_tensorcnt(0);
    __syncthreads();

    if (kb <= q0 + 15) {   // tile not fully masked for this wave
      // --- S = Q * K^T : 2 N-tiles x 2 K-steps ---
      v8f s[2];
      s[0] = 0.0f; s[1] = 0.0f;
#pragma unroll
      for (int nt = 0; nt < 2; ++nt) {
#pragma unroll
        for (int ks = 0; ks < 2; ++ks) {
          FragA bfr;
          const int krow = nt * 16 + lhalf;
          const int ko   = ks * 32 + hi * 8;
          bfr.u[0] = *(const ushort8*)&Kt[krow * 64 + ko];
          bfr.u[1] = *(const ushort8*)&Kt[krow * 64 + ko + 16];
          s[nt] = __builtin_amdgcn_wmma_f32_16x16x32_bf16(
              false, qf[ks].bf, false, bfr.bf, (short)0, s[nt], false, false);
        }
      }
      // --- online softmax (rows live in 16-lane halves) ---
#pragma unroll
      for (int r = 0; r < 8; ++r) {
        const int qrow = q0 + r + hi * 8;
        float s0 = s[0][r] * 0.125f;          // 1/sqrt(64)
        float s1 = s[1][r] * 0.125f;
        if (kb + lhalf > qrow)      s0 = -1e30f;
        if (kb + 16 + lhalf > qrow) s1 = -1e30f;
        float vmax = fmaxf(s0, s1);
#pragma unroll
        for (int off = 8; off >= 1; off >>= 1)
          vmax = fmaxf(vmax, __shfl_xor(vmax, off, 32));
        const float mnew  = fmaxf(mr[r], vmax);
        const float scale = __expf(mr[r] - mnew);
        mr[r] = mnew;
        const float p0 = __expf(s0 - mnew);
        const float p1 = __expf(s1 - mnew);
        float ps = p0 + p1;
#pragma unroll
        for (int off = 8; off >= 1; off >>= 1)
          ps += __shfl_xor(ps, off, 32);
        sr[r] = sr[r] * scale + ps;
#pragma unroll
        for (int dt = 0; dt < 4; ++dt) o[dt][r] *= scale;
        const int prow = r + hi * 8;
        pw[prow * 32 + lhalf]      = f2bf(p0);
        pw[prow * 32 + 16 + lhalf] = f2bf(p1);
      }
      // --- O += P * V : re-layout P via LDS, V from transposed tile ---
      FragA pf;
      pf.u[0] = *(const ushort8*)&pw[lhalf * 32 + hi * 8];
      pf.u[1] = *(const ushort8*)&pw[lhalf * 32 + hi * 8 + 16];
#pragma unroll
      for (int dt = 0; dt < 4; ++dt) {
        FragA vb;
        const int d = dt * 16 + lhalf;
        vb.u[0] = *(const ushort8*)&Vt[d * 40 + hi * 8];
        vb.u[1] = *(const ushort8*)&Vt[d * 40 + hi * 8 + 16];
        o[dt] = __builtin_amdgcn_wmma_f32_16x16x32_bf16(
            false, pf.bf, false, vb.bf, (short)0, o[dt], false, false);
      }
    }
    __syncthreads();
  }

  // --- normalize and store y (bf16, [B,T,C] with head offset) ---
  const int b = bh >> 4;
  const int h = bh & 15;
#pragma unroll
  for (int r = 0; r < 8; ++r) {
    const int t = q0 + r + hi * 8;
    const float inv = 1.0f / sr[r];
#pragma unroll
    for (int dt = 0; dt < 4; ++dt) {
      const int c = h * HD + dt * 16 + lhalf;
      Yg[(size_t)(b * TVAL + t) * CVAL + c] = f2bf(o[dt][r] * inv);
    }
  }
}

// ---------------------------------------------------------------- launch
extern "C" void kernel_launch(void* const* d_in, const int* in_sizes, int n_in,
                              void* d_out, int out_size, void* d_ws,
                              size_t ws_size, hipStream_t stream) {
  const float* x      = (const float*)d_in[0];
  const float* W_attn = (const float*)d_in[1];
  const float* b_attn = (const float*)d_in[2];
  const float* W_proj = (const float*)d_in[3];
  const float* b_proj = (const float*)d_in[4];
  float* out = (float*)d_out;

  // workspace carve (all bf16 as u16); total 48 MiB
  u16* Xbf = (u16*)d_ws;
  u16* Wa  = Xbf + (size_t)BTOT * CVAL;          // x in bf16
  u16* Wp  = Wa + (size_t)3 * CVAL * CVAL;       // W_attn bf16
  u16* Qb  = Wp + (size_t)CVAL * CVAL;           // W_proj bf16
  u16* Kb  = Qb + (size_t)BTOT * CVAL;           // q [B,H,T,hd]
  u16* Vb  = Kb + (size_t)BTOT * CVAL;           // k
  u16* Ybf = Vb + (size_t)BTOT * CVAL;           // v ; then y bf16 [B,T,C]

  f32_to_bf16_kernel<<<2048, 256, 0, stream>>>(x, Xbf, BTOT * CVAL);
  f32_to_bf16_kernel<<<2048, 256, 0, stream>>>(W_attn, Wa, 3 * CVAL * CVAL);
  f32_to_bf16_kernel<<<1024, 256, 0, stream>>>(W_proj, Wp, CVAL * CVAL);

  dim3 gQKV(3 * CVAL / 64, BTOT / 64);   // 48 x 64
  gemm_bf16_wmma<1><<<gQKV, 128, 0, stream>>>(Xbf, Wa, b_attn, nullptr, Qb,
                                              Kb, Vb, BTOT, 3 * CVAL, CVAL);

  dim3 gAtt(TVAL / 64, 2 * HVAL);        // 32 x 32
  attn_wmma<<<gAtt, 128, 0, stream>>>(Qb, Kb, Vb, Ybf);

  dim3 gPrj(CVAL / 64, BTOT / 64);       // 16 x 64
  gemm_bf16_wmma<0><<<gPrj, 128, 0, stream>>>(Ybf, Wp, b_proj, out, nullptr,
                                              nullptr, nullptr, BTOT, CVAL,
                                              CVAL);
}
